// Attention_28003186770620
// MI455X (gfx1250) — compile-verified, matrix operand order checked by reading, not run on hardware
//
#include <hip/hip_runtime.h>

// ---------- CDNA5 WMMA types ----------
typedef __attribute__((ext_vector_type(16))) __bf16 v16bf;
typedef __attribute__((ext_vector_type(8)))  float  v8f;

union FragU { uint4 u4[2]; v16bf v; };

// hardware tanh if available (V_TANH_F32 on gfx1250)
#if __has_builtin(__builtin_amdgcn_tanhf)
#define TANH(x) __builtin_amdgcn_tanhf(x)
#elif __has_builtin(__builtin_amdgcn_tanh_f32)
#define TANH(x) __builtin_amdgcn_tanh_f32(x)
#else
#define TANH(x) tanhf(x)
#endif

// round-to-nearest-even f32 -> bf16 (packed pair)
__device__ __forceinline__ unsigned int f2bf(float f) {
  unsigned int u = __float_as_uint(f);
  return (u + 0x7FFFu + ((u >> 16) & 1u)) >> 16;
}
__device__ __forceinline__ unsigned int bfpk(float lo, float hi) {
  return f2bf(lo) | (f2bf(hi) << 16);
}

#define B_DIM 32
#define L_DIM 2048
#define H_DIM 1024
// LDS keys-tile layout: 16 lane-groups (l15), each 8 M-slots of 516 dwords,
// group stride padded to 4132 dwords -> conflict-free b128, all offsets < 64KB.
#define MSTRIDE 516u
#define GSTRIDE 4132u

// ---------------------------------------------------------------------------
// k0: convert ua_w (f32 [1024,1024]) -> packed bf16 pairs in workspace
// ---------------------------------------------------------------------------
__global__ __launch_bounds__(256) void cvt_uaw_kernel(
    const float* __restrict__ ua_w, unsigned int* __restrict__ dst) {
  size_t i = ((size_t)blockIdx.x * 256 + threadIdx.x) * 8;
  float4 a = *(const float4*)(ua_w + i);
  float4 c = *(const float4*)(ua_w + i + 4);
  uint4 o;
  o.x = bfpk(a.x, a.y); o.y = bfpk(a.z, a.w);
  o.z = bfpk(c.x, c.y); o.w = bfpk(c.z, c.w);
  *(uint4*)(dst + (i >> 1)) = o;
}

// ---------------------------------------------------------------------------
// k1: pre[b,k] = wa_b[k] + ua_b[k] + dot(query[b,:], wa_w[k,:])
// ---------------------------------------------------------------------------
__global__ __launch_bounds__(256) void pre_kernel(
    const float* __restrict__ query, const float* __restrict__ wa_w,
    const float* __restrict__ wa_b, const float* __restrict__ ua_b,
    float* __restrict__ pre) {
  int b = blockIdx.y;
  int k = blockIdx.x * 256 + threadIdx.x;
  const float4* q = (const float4*)(query + (size_t)b * H_DIM);
  const float4* w = (const float4*)(wa_w + (size_t)k * H_DIM);
  float4 a = {0.f, 0.f, 0.f, 0.f};
  for (int i = 0; i < H_DIM / 4; ++i) {
    float4 qa = q[i], wa = w[i];
    a.x += qa.x * wa.x; a.y += qa.y * wa.y;
    a.z += qa.z * wa.z; a.w += qa.w * wa.w;
  }
  pre[(size_t)b * H_DIM + k] = wa_b[k] + ua_b[k] + a.x + a.y + a.z + a.w;
}

// ---------------------------------------------------------------------------
// helpers for the software-pipelined GEMM core
// ---------------------------------------------------------------------------
__device__ __forceinline__ void load_b(FragU& f, const unsigned int* brow, int kc) {
  f.u4[0] = *(const uint4*)(brow + kc * 16);       // global_load_b128 (L2)
  f.u4[1] = *(const uint4*)(brow + kc * 16 + 8);
}

// 8 WMMAs for one K-chunk, A fragments double-buffered ahead of each WMMA.
__device__ __forceinline__ void mstep(v8f* acc, const unsigned int* abase,
                                      int kc, const FragU& bf) {
  FragU a0, a1;
  a0.u4[0] = *(const uint4*)(abase + kc * 16);
  a0.u4[1] = *(const uint4*)(abase + kc * 16 + 8);
  #pragma unroll
  for (int m = 0; m < 8; ++m) {
    FragU& cur = (m & 1) ? a1 : a0;
    FragU& nxt = (m & 1) ? a0 : a1;
    if (m < 7) {  // folds statically under full unroll
      nxt.u4[0] = *(const uint4*)(abase + (m + 1) * MSTRIDE + kc * 16);
      nxt.u4[1] = *(const uint4*)(abase + (m + 1) * MSTRIDE + kc * 16 + 8);
    }
    acc[m] = __builtin_amdgcn_wmma_f32_16x16x32_bf16(
        false, cur.v, false, bf.v, (short)0, acc[m], false, false);
  }
}

// ---------------------------------------------------------------------------
// k2: fused scores kernel.
// grid (L/128, B), block 256 (8 wave32s), dynamic LDS ~267 KB of 320 KB/WGP.
// scores[b,l] = sum_n va_w[n] * tanh(pre[b,n] + dot(keys[b,l,:], ua_w[n,:]))
// ---------------------------------------------------------------------------
__global__ __launch_bounds__(256) void score_kernel(
    const float* __restrict__ keys, const unsigned int* __restrict__ uawb,
    const float* __restrict__ pre, const float* __restrict__ vaw,
    float* __restrict__ scores) {
  extern __shared__ __align__(16) char smem[];
  unsigned int* kl = (unsigned int*)smem;                    // 16*GSTRIDE dwords
  float* s_pre = (float*)(smem + 16 * GSTRIDE * 4);          // 1024
  float* s_va  = s_pre + H_DIM;                              // 1024
  float* s_sc  = s_va + H_DIM;                               // 128

  const int b  = blockIdx.y;
  const int l0 = blockIdx.x * 128;
  const int t  = threadIdx.x;

  // Async-stage pre[b,:] and va_w straight into LDS (CDNA5 async DMA path);
  // runs underneath the keys-tile staging below.
  {
    unsigned lp = (unsigned)(uintptr_t)(s_pre + t * 4);
    unsigned lv = (unsigned)(uintptr_t)(s_va + t * 4);
    const float* gp = pre + (size_t)b * H_DIM + t * 4;
    const float* gv = vaw + t * 4;
    asm volatile("global_load_async_to_lds_b128 %0, %1, off"
                 :: "v"(lp), "v"(gp) : "memory");
    asm volatile("global_load_async_to_lds_b128 %0, %1, off"
                 :: "v"(lv), "v"(gv) : "memory");
  }
  if (t < 128) s_sc[t] = 0.0f;

  // Stage keys tile (f32 -> bf16) into LDS, lane-group permuted layout:
  // global row g = m*16 + l15  ->  LDS offset l15*GSTRIDE + m*MSTRIDE.
  const float4* kg = (const float4*)(keys + ((size_t)b * L_DIM + l0) * H_DIM);
  for (int it = 0; it < 128; ++it) {
    int idx = it * 256 + t;
    int row = idx >> 8;          // 0..127 (one row per it)
    int c4  = idx & 255;         // float4 column
    float4 v = kg[(size_t)row * 256 + c4];
    uint2 p; p.x = bfpk(v.x, v.y); p.y = bfpk(v.z, v.w);
    unsigned off = (unsigned)(row & 15) * GSTRIDE + (unsigned)(row >> 4) * MSTRIDE + c4 * 2;
    *(uint2*)(kl + off) = p;     // ds_store_b64, conflict-free
  }
  asm volatile("s_wait_asynccnt 0x0" ::: "memory");
  __syncthreads();

  const int lane = t & 31;
  const int wv   = t >> 5;        // wave id 0..7
  const int half = lane >> 4;     // WMMA lane-half
  const int l15  = lane & 15;

  float srow[64];
  #pragma unroll
  for (int i = 0; i < 64; ++i) srow[i] = 0.f;

  // All A-fragment addresses = abase + compile-time constant (DS imm offsets).
  const unsigned int* abase = kl + l15 * GSTRIDE + half * 4;

  // Each wave handles 8 of the 64 N-tiles; B fragment reused across 8 M-tiles.
  for (int nti = 0; nti < 8; ++nti) {
    const int n0 = (wv + nti * 8) * 16;
    const float pre_n = s_pre[n0 + l15];
    const float va_n  = s_va [n0 + l15];

    v8f acc[8] = {};
    const unsigned int* brow = uawb + (size_t)(n0 + l15) * (H_DIM / 2) + half * 4;
    if (nti < 7) {  // warm the L2 path for next tile's B stream
      __builtin_prefetch(uawb + (size_t)(n0 + 128 + l15) * (H_DIM / 2), 0, 1);
    }

    // Software-pipelined over kc: B(kc+1) is issued before the 8 WMMAs of kc,
    // so the L2 round-trip is covered by a full chunk of matrix work.
    FragU bfA, bfB;
    load_b(bfA, brow, 0);
    #pragma unroll 2
    for (int kc = 0; kc < 32; kc += 2) {
      load_b(bfB, brow, kc + 1);
      mstep(acc, abase, kc, bfA);
      load_b(bfA, brow, (kc + 2) & 31);   // wraps at the end (2 redundant loads)
      mstep(acc, abase, kc + 1, bfB);
    }

    // epilogue: tanh + va_w weighting, accumulate per-row partials in regs
    #pragma unroll
    for (int m = 0; m < 8; ++m) {
      #pragma unroll
      for (int r = 0; r < 8; ++r) {
        srow[m * 8 + r] += TANH(acc[m][r] + pre_n) * va_n;
      }
    }
  }

  // reduce the 16 columns across lanes, then merge waves via LDS atomics
  #pragma unroll
  for (int m = 0; m < 8; ++m) {
    #pragma unroll
    for (int r = 0; r < 8; ++r) {
      float v = srow[m * 8 + r];
      v += __shfl_xor(v, 1, 32);
      v += __shfl_xor(v, 2, 32);
      v += __shfl_xor(v, 4, 32);
      v += __shfl_xor(v, 8, 32);
      if (l15 == 0) atomicAdd(&s_sc[m * 16 + half * 8 + r], v);
    }
  }
  __syncthreads();
  if (t < 128) scores[(size_t)b * L_DIM + l0 + t] = s_sc[t];
}

// ---------------------------------------------------------------------------
// k3: masked softmax over L per batch. grid 32, block 256.
// (va_b omitted: softmax is shift-invariant.)
// ---------------------------------------------------------------------------
__global__ __launch_bounds__(256) void softmax_kernel(
    const float* __restrict__ scores, const unsigned char* __restrict__ mask,
    float* __restrict__ weights) {
  __shared__ float red[256];
  int b = blockIdx.x, t = threadIdx.x;
  float vals[8];
  float mx = -__builtin_inff();
  #pragma unroll
  for (int i = 0; i < 8; ++i) {
    int l = t + i * 256;
    float s = scores[(size_t)b * L_DIM + l];
    if (mask[(size_t)b * L_DIM + l]) s = -__builtin_inff();
    vals[i] = s;
    mx = fmaxf(mx, s);
  }
  red[t] = mx; __syncthreads();
  for (int o = 128; o > 0; o >>= 1) { if (t < o) red[t] = fmaxf(red[t], red[t + o]); __syncthreads(); }
  mx = red[0]; __syncthreads();

  float sum = 0.f;
  #pragma unroll
  for (int i = 0; i < 8; ++i) { vals[i] = __expf(vals[i] - mx); sum += vals[i]; }
  red[t] = sum; __syncthreads();
  for (int o = 128; o > 0; o >>= 1) { if (t < o) red[t] += red[t + o]; __syncthreads(); }
  float inv = 1.0f / red[0];
  #pragma unroll
  for (int i = 0; i < 8; ++i)
    weights[(size_t)b * L_DIM + t + i * 256] = vals[i] * inv;
}

// ---------------------------------------------------------------------------
// k4: context[b,h] = sum_l weights[b,l] * keys[b,l,h]. grid (8, 32), block 256.
// ---------------------------------------------------------------------------
__global__ __launch_bounds__(256) void context_kernel(
    const float* __restrict__ weights, const float* __restrict__ keys,
    float* __restrict__ out) {
  __shared__ float  w_s[L_DIM];
  __shared__ float4 red[256];
  int b = blockIdx.y, h0 = blockIdx.x * 128, t = threadIdx.x;
  for (int i = t; i < L_DIM; i += 256) w_s[i] = weights[(size_t)b * L_DIM + i];
  __syncthreads();

  int c4 = t & 31;   // float4 column within 128-wide chunk
  int lr = t >> 5;   // 0..7 row-phase
  float4 acc = {0.f, 0.f, 0.f, 0.f};
  const float4* kb = (const float4*)(keys + (size_t)b * L_DIM * H_DIM + h0);
  for (int l = lr; l < L_DIM; l += 8) {
    float w = w_s[l];
    float4 kv = kb[(size_t)l * 256 + c4];
    acc.x += w * kv.x; acc.y += w * kv.y; acc.z += w * kv.z; acc.w += w * kv.w;
  }
  red[t] = acc; __syncthreads();
  if (lr == 0) {
    float4 s = red[t];
    for (int j = 1; j < 8; ++j) {
      float4 o = red[t + j * 32];
      s.x += o.x; s.y += o.y; s.z += o.z; s.w += o.w;
    }
    *(float4*)(out + (size_t)b * H_DIM + h0 + c4 * 4) = s;
  }
}

// ---------------------------------------------------------------------------
extern "C" void kernel_launch(void* const* d_in, const int* in_sizes, int n_in,
                              void* d_out, int out_size, void* d_ws, size_t ws_size,
                              hipStream_t stream) {
  const float* query = (const float*)d_in[0];
  const float* keys  = (const float*)d_in[1];
  const unsigned char* mask = (const unsigned char*)d_in[2];
  const float* wa_w = (const float*)d_in[3];
  const float* wa_b = (const float*)d_in[4];
  const float* ua_w = (const float*)d_in[5];
  const float* ua_b = (const float*)d_in[6];
  const float* va_w = (const float*)d_in[7];
  // d_in[8] = va_b: dropped (softmax shift-invariance)
  float* out = (float*)d_out;

  char* ws = (char*)d_ws;
  unsigned int* uawb = (unsigned int*)ws;                         // 2 MB  bf16 ua_w
  float* pre     = (float*)(ws + (2u << 20));                     // 128 KB
  float* scores  = (float*)(ws + (2u << 20) + (128u << 10));      // 256 KB
  float* weights = (float*)(ws + (2u << 20) + (384u << 10));      // 256 KB

  cvt_uaw_kernel<<<512, 256, 0, stream>>>(ua_w, uawb);
  pre_kernel<<<dim3(4, B_DIM), 256, 0, stream>>>(query, wa_w, wa_b, ua_b, pre);

  size_t smem = (size_t)16 * GSTRIDE * 4 + H_DIM * 4 + H_DIM * 4 + 128 * 4;
  score_kernel<<<dim3(L_DIM / 128, B_DIM), 256, smem, stream>>>(keys, uawb, pre, va_w, scores);

  softmax_kernel<<<B_DIM, 256, 0, stream>>>(scores, mask, weights);
  context_kernel<<<dim3(H_DIM / 128, B_DIM), 256, 0, stream>>>(weights, keys, out);
}
